// Attention_80075370266831
// MI455X (gfx1250) — matrix operand order repaired and checked by reading, not compiled, not run on hardware
//
#include <hip/hip_runtime.h>
#include <hip/hip_bf16.h>

// ---------- WMMA fragment types (CDNA5 gfx1250, wave32) ----------
typedef __attribute__((ext_vector_type(16))) __bf16 v16bf;
typedef __attribute__((ext_vector_type(8)))  float  v8f;

union Frag { uint4 q[2]; v16bf v; };

#define WMMA_BF16(A, B, C) \
  __builtin_amdgcn_wmma_f32_16x16x32_bf16(false, (A), false, (B), (short)0, (C), false, false)

__device__ __forceinline__ unsigned short f2bf(float f) {
  unsigned u = __float_as_uint(f);
  u += 0x7FFFu + ((u >> 16) & 1u);   // round-to-nearest-even
  return (unsigned short)(u >> 16);
}
__device__ __forceinline__ float bf2f(unsigned short h) {
  return __uint_as_float(((unsigned)h) << 16);
}

// =====================================================================
// Generic GEMM:  C[M,Nout] = A[M,K] * W[K,Nout]
//   A: fp32 or bf16 (template).  W: fp32 (converted to bf16 on stage).
//   Output: bf16, or fp32 with bias (template).
// Block: 256 threads = 8 waves, tile 128(M) x 128(N), K-step 32.
// Double-buffered LDS software pipeline (one barrier per k-step) +
// ping-pong B-fragment registers so LDS frag loads overlap WMMAs.
// Requires K % 32 == 0 and Nout % 128 == 0. M edge: clamp + store guard.
// =====================================================================
template <bool A_BF16, bool BIAS_OUT_F32>
__global__ __launch_bounds__(256)
void gemm_bf16_wmma(const void* __restrict__ Ap, const float* __restrict__ W,
                    const float* __restrict__ bias, void* __restrict__ Cp,
                    int M, int K, int Nout)
{
  __shared__ __align__(16) unsigned short sA[2][128 * 40];  // [buf][row][k]
  __shared__ __align__(16) unsigned short sB[2][128 * 40];  // [buf][col][k]

  const int tid  = threadIdx.x;
  const int lane = tid & 31;
  const int wv   = tid >> 5;        // 0..7
  const int half = lane >> 4;       // 0/1
  const int l16  = lane & 15;

  const int m0 = blockIdx.x * 128;
  const int n0 = blockIdx.y * 128;

  v8f acc[8] = {};

  const int arow = tid >> 1;            // 0..127
  const int acol = (tid & 1) * 16;      // 0 or 16
  const int bkk  = tid >> 3;            // 0..31
  const int bnb  = (tid & 7) * 16;      // 0..112

  // clamp A row (garbage rows only feed unstored outputs; avoids OOB reads)
  int garow = m0 + arow; if (garow > M - 1) garow = M - 1;

  unsigned short aReg[16];
  float4 wReg[4];

  auto stageLoad = [&](int k0) {
    if (A_BF16) {
      const unsigned short* ap =
          (const unsigned short*)Ap + (size_t)garow * K + k0 + acol;
      *(uint4*)&aReg[0] = *(const uint4*)ap;
      *(uint4*)&aReg[8] = *(const uint4*)(ap + 8);
    } else {
      const float4* ap4 =
          (const float4*)((const float*)Ap + (size_t)garow * K + k0 + acol);
      #pragma unroll
      for (int j = 0; j < 4; ++j) {
        float4 f = ap4[j];
        aReg[4 * j + 0] = f2bf(f.x); aReg[4 * j + 1] = f2bf(f.y);
        aReg[4 * j + 2] = f2bf(f.z); aReg[4 * j + 3] = f2bf(f.w);
      }
    }
    const float4* wp4 =
        (const float4*)(W + (size_t)(k0 + bkk) * Nout + n0 + bnb);
    #pragma unroll
    for (int j = 0; j < 4; ++j) wReg[j] = wp4[j];
  };

  auto stageStore = [&](int buf) {
    *(uint4*)&sA[buf][arow * 40 + acol]     = *(uint4*)&aReg[0];
    *(uint4*)&sA[buf][arow * 40 + acol + 8] = *(uint4*)&aReg[8];
    #pragma unroll
    for (int j = 0; j < 4; ++j) {
      sB[buf][(bnb + 4 * j + 0) * 40 + bkk] = f2bf(wReg[j].x);
      sB[buf][(bnb + 4 * j + 1) * 40 + bkk] = f2bf(wReg[j].y);
      sB[buf][(bnb + 4 * j + 2) * 40 + bkk] = f2bf(wReg[j].z);
      sB[buf][(bnb + 4 * j + 3) * 40 + bkk] = f2bf(wReg[j].w);
    }
  };

  // ---- pipeline prologue ----
  stageLoad(0);
  stageStore(0);
  __syncthreads();

  const int NK = K >> 5;
  for (int i = 0; i < NK; ++i) {
    const int cur = i & 1;
    if (i + 1 < NK) stageLoad(32 * (i + 1));      // global loads overlap WMMAs

    Frag fa;
    const unsigned short* arp = &sA[cur][(16 * wv + l16) * 40];
    fa.q[0] = *(const uint4*)(arp + (half ? 8 : 0));
    fa.q[1] = *(const uint4*)(arp + 16 + (half ? 8 : 0));

    // ping-pong B fragments: load j+1 while WMMA consumes j
    Frag fb[2];
    auto loadB = [&](int j, Frag& f) {
      const unsigned short* brp = &sB[cur][(16 * j + l16) * 40 + (half ? 8 : 0)];
      f.q[0] = *(const uint4*)brp;
      f.q[1] = *(const uint4*)(brp + 16);
    };
    loadB(0, fb[0]);
    #pragma unroll
    for (int j = 0; j < 8; ++j) {
      if (j + 1 < 8) loadB(j + 1, fb[(j + 1) & 1]);
      acc[j] = WMMA_BF16(fa.v, fb[j & 1].v, acc[j]);
    }

    if (i + 1 < NK) stageStore(cur ^ 1);
    __syncthreads();
  }

  // ---- epilogue ----
  #pragma unroll
  for (int j = 0; j < 8; ++j) {
    const int col = n0 + 16 * j + l16;
    #pragma unroll
    for (int r = 0; r < 8; ++r) {
      const int grow = m0 + 16 * wv + r + 8 * half;
      if (grow < M) {
        if (BIAS_OUT_F32)
          ((float*)Cp)[(size_t)grow * Nout + col] = acc[j][r] + bias[col];
        else
          ((unsigned short*)Cp)[(size_t)grow * Nout + col] = f2bf(acc[j][r]);
      }
    }
  }
}

// =====================================================================
// v2[b,h,n,d] = v[b,h,n,d] - sum_m s[h,n,m] * v[b,h,m,d]
// K-loop split: 18 full steps (cols <= 575, no clamping at all) + one
// peeled tail step (m0 = 576, 1 valid column; A broadcast-clamped, B
// K-rows >= 577 zeroed which makes A padding values irrelevant).
// =====================================================================
__global__ __launch_bounds__(256)
void v2_kernel(const float* __restrict__ s, const unsigned short* __restrict__ qkv,
               unsigned short* __restrict__ v2)
{
  __shared__ __align__(16) unsigned short sA[128 * 40];
  __shared__ __align__(16) unsigned short sB[64 * 40];

  const int b  = blockIdx.y >> 4;
  const int h  = blockIdx.y & 15;
  const int n0 = blockIdx.x * 128;

  const int tid  = threadIdx.x;
  const int lane = tid & 31;
  const int wv   = tid >> 5;
  const int half = lane >> 4;
  const int l16  = lane & 15;

  const float* sh = s + (size_t)h * 577 * 577;
  v8f acc[4] = {};

  const int arow = tid >> 1;
  const int acol = (tid & 1) * 16;
  const int bkk  = tid >> 3;
  const int bnb  = (tid & 7) * 8;

  // clamp A row (s rows >= 577 only feed unstored output rows)
  int gnc = n0 + arow; if (gnc > 576) gnc = 576;
  const float* ap = sh + (size_t)gnc * 577;

  auto compute_step = [&]() {
    Frag fa;
    const unsigned short* arp = &sA[(16 * wv + l16) * 40];
    fa.q[0] = *(const uint4*)(arp + (half ? 8 : 0));
    fa.q[1] = *(const uint4*)(arp + 16 + (half ? 8 : 0));
    Frag fb[2];
    auto loadB = [&](int j, Frag& f) {
      const unsigned short* brp = &sB[(16 * j + l16) * 40 + (half ? 8 : 0)];
      f.q[0] = *(const uint4*)brp;
      f.q[1] = *(const uint4*)(brp + 16);
    };
    loadB(0, fb[0]);
    #pragma unroll
    for (int j = 0; j < 4; ++j) {
      if (j + 1 < 4) loadB(j + 1, fb[(j + 1) & 1]);
      acc[j] = WMMA_BF16(fa.v, fb[j & 1].v, acc[j]);
    }
  };

  // ---- 18 full steps: no clamps anywhere ----
  for (int m0 = 0; m0 < 576; m0 += 32) {
    {
      const float* base = ap + m0 + acol;     // one base, immediate offsets
      unsigned short tmp[16];
      #pragma unroll
      for (int j = 0; j < 16; ++j) tmp[j] = f2bf(base[j]);
      *(uint4*)&sA[arow * 40 + acol]     = *(uint4*)&tmp[0];
      *(uint4*)&sA[arow * 40 + acol + 8] = *(uint4*)&tmp[8];
    }
    {
      const unsigned short* vp =
          qkv + ((size_t)(b * 577 + m0 + bkk)) * 3072 + 2048 + h * 64 + bnb;
      unsigned short tb[8];
      *(uint4*)&tb[0] = *(const uint4*)vp;
      #pragma unroll
      for (int j = 0; j < 8; ++j) sB[(bnb + j) * 40 + bkk] = tb[j];
    }
    __syncthreads();
    compute_step();
    __syncthreads();
  }

  // ---- peeled tail: m0 = 576 (cols 576..607, only 576 real) ----
  {
    const unsigned short abb = f2bf(ap[576]);   // broadcast (B zeroes the pad)
    unsigned short tmp[16];
    #pragma unroll
    for (int j = 0; j < 16; ++j) tmp[j] = abb;
    *(uint4*)&sA[arow * 40 + acol]     = *(uint4*)&tmp[0];
    *(uint4*)&sA[arow * 40 + acol + 8] = *(uint4*)&tmp[8];

    const unsigned short* vp =
        qkv + ((size_t)(b * 577 + 576)) * 3072 + 2048 + h * 64 + bnb;
    unsigned short tb[8];
    *(uint4*)&tb[0] = *(const uint4*)vp;
    #pragma unroll
    for (int j = 0; j < 8; ++j)
      sB[(bnb + j) * 40 + bkk] = (bkk == 0) ? tb[j] : (unsigned short)0;
    __syncthreads();
    compute_step();
  }

  // ---- epilogue: v2 = v - acc ----
  #pragma unroll
  for (int j = 0; j < 4; ++j) {
    const int d = 16 * j + l16;
    #pragma unroll
    for (int r = 0; r < 8; ++r) {
      const int gn = n0 + 16 * wv + r + 8 * half;
      if (gn < 577) {
        const float vv =
            bf2f(qkv[((size_t)(b * 577 + gn)) * 3072 + 2048 + h * 64 + d]);
        v2[(((size_t)(b * 16 + h)) * 577 + gn) * 64 + d] = f2bf(vv - acc[j][r]);
      }
    }
  }
}

// =====================================================================
// Attention: one 16-query tile per 128-thread workgroup (4 waves).
//  pass1: S = Q K^T (WMMA), scaled, padded cols -> -inf
//  pass2: row softmax in LDS, bf16 P
//  pass3: O = P V2 (WMMA, V2 chunks transposed through LDS)
// =====================================================================
__global__ __launch_bounds__(128)
void attn_kernel(const unsigned short* __restrict__ qkv,
                 const unsigned short* __restrict__ v2,
                 unsigned short* __restrict__ ctx)
{
  __shared__ __align__(16) float          Sc[16 * 608];
  __shared__ __align__(16) unsigned short Pl[16 * 608];
  __shared__ __align__(16) unsigned short sV[64 * 40];
  __shared__ float red[16][8];

  const int nq0 = blockIdx.x * 16;
  const int b   = blockIdx.y;
  const int h   = blockIdx.z;

  const int tid  = threadIdx.x;
  const int lane = tid & 31;
  const int wv   = tid >> 5;       // 0..3
  const int half = lane >> 4;
  const int l16  = lane & 15;

  const float SCALE = 0.125f;      // 64^-0.5

  // ---- Q fragments (16 rows x 64 dims = two K=32 chunks), direct from global
  int qrow = nq0 + l16; if (qrow > 576) qrow = 576;
  const unsigned short* qp = qkv + ((size_t)(b * 577 + qrow)) * 3072 + h * 64;
  Frag qa0, qa1;
  qa0.q[0] = *(const uint4*)(qp + (half ? 8 : 0));
  qa0.q[1] = *(const uint4*)(qp + 16 + (half ? 8 : 0));
  qa1.q[0] = *(const uint4*)(qp + 32 + (half ? 8 : 0));
  qa1.q[1] = *(const uint4*)(qp + 48 + (half ? 8 : 0));

  // ---- pass 1: scores, 38 key tiles of 16 (608 padded) ----
  for (int t = wv; t < 38; t += 4) {
    const int kt0 = t * 16;
    int key = kt0 + l16; if (key > 576) key = 576;
    const unsigned short* kp =
        qkv + ((size_t)(b * 577 + key)) * 3072 + 1024 + h * 64 + (half ? 8 : 0);
    Frag kb0, kb1;
    kb0.q[0] = *(const uint4*)(kp);
    kb0.q[1] = *(const uint4*)(kp + 16);
    kb1.q[0] = *(const uint4*)(kp + 32);
    kb1.q[1] = *(const uint4*)(kp + 48);
    v8f sa = {};
    sa = WMMA_BF16(qa0.v, kb0.v, sa);
    sa = WMMA_BF16(qa1.v, kb1.v, sa);
    const int colg = kt0 + l16;
    #pragma unroll
    for (int r = 0; r < 8; ++r) {
      const int row = r + 8 * half;
      Sc[row * 608 + colg] = (colg < 577) ? sa[r] * SCALE : -1e30f;
    }
  }
  __syncthreads();

  // ---- pass 2: softmax (row r handled by 8 threads) ----
  const int rr = tid >> 3, ss = tid & 7;
  float mx = -3e30f;
  for (int c = ss; c < 608; c += 8) mx = fmaxf(mx, Sc[rr * 608 + c]);
  red[rr][ss] = mx;
  __syncthreads();
  float rm = red[rr][0];
  #pragma unroll
  for (int j = 1; j < 8; ++j) rm = fmaxf(rm, red[rr][j]);
  __syncthreads();
  float sum = 0.f;
  for (int c = ss; c < 608; c += 8) {
    float e = __expf(Sc[rr * 608 + c] - rm);
    Sc[rr * 608 + c] = e;
    sum += e;
  }
  red[rr][ss] = sum;
  __syncthreads();
  float rs = 0.f;
  #pragma unroll
  for (int j = 0; j < 8; ++j) rs += red[rr][j];
  const float inv = 1.0f / rs;
  for (int c = ss; c < 608; c += 8) Pl[rr * 608 + c] = f2bf(Sc[rr * 608 + c] * inv);
  __syncthreads();

  // ---- pass 3: O = P (16x608) * V2 (608x64); wave wv owns cols 16*wv..+15 ----
  v8f oacc = {};
  const int vml = tid >> 2;          // 0..31 (m within chunk)
  const int vdb = (tid & 3) * 16;    // d base
  const unsigned short* vbase = v2 + ((size_t)(b * 16 + h)) * 577 * 64;
  for (int m0 = 0; m0 < 608; m0 += 32) {
    unsigned short t16[16];
    if (m0 < 576) {                                   // uniform fast path
      const unsigned short* vp = vbase + (size_t)(m0 + vml) * 64 + vdb;
      *(uint4*)&t16[0] = *(const uint4*)vp;
      *(uint4*)&t16[8] = *(const uint4*)(vp + 8);
    } else {                                          // tail: only m=576 real
      const unsigned short* vp = vbase + (size_t)576 * 64 + vdb;
      *(uint4*)&t16[0] = *(const uint4*)vp;
      *(uint4*)&t16[8] = *(const uint4*)(vp + 8);
      #pragma unroll
      for (int j = 0; j < 16; ++j)
        t16[j] = (vml == 0) ? t16[j] : (unsigned short)0;  // zero K-pad
    }
    #pragma unroll
    for (int j = 0; j < 16; ++j) sV[(vdb + j) * 40 + vml] = t16[j];
    __syncthreads();

    Frag pa, vb;
    const unsigned short* pp = Pl + l16 * 608 + m0;
    pa.q[0] = *(const uint4*)(pp + (half ? 8 : 0));
    pa.q[1] = *(const uint4*)(pp + 16 + (half ? 8 : 0));
    const unsigned short* vbp = sV + (16 * wv + l16) * 40 + (half ? 8 : 0);
    vb.q[0] = *(const uint4*)vbp;
    vb.q[1] = *(const uint4*)(vbp + 16);
    oacc = WMMA_BF16(pa.v, vb.v, oacc);
    __syncthreads();
  }

  // ---- store merged-heads ctx[b, n, h*64 + d] as bf16 ----
  const int d = h * 64 + 16 * wv + l16;
  #pragma unroll
  for (int r = 0; r < 8; ++r) {
    const int gn = nq0 + r + 8 * half;
    if (gn < 577)
      ctx[((size_t)(b * 577) + gn) * 1024 + d] = f2bf(oacc[r]);
  }
}

// =====================================================================
extern "C" void kernel_launch(void* const* d_in, const int* in_sizes, int n_in,
                              void* d_out, int out_size, void* d_ws, size_t ws_size,
                              hipStream_t stream)
{
  (void)in_sizes; (void)n_in; (void)out_size; (void)ws_size;
  const float* x     = (const float*)d_in[0];
  const float* Wqkv  = (const float*)d_in[1];
  const float* Wproj = (const float*)d_in[2];
  const float* bproj = (const float*)d_in[3];
  const float* s     = (const float*)d_in[4];

  const int B = 32, N = 577, C = 1024, H = 16, D = 64;
  const int M = B * N;                       // 18464

  unsigned short* qkv = (unsigned short*)d_ws;              // [M, 3072] bf16
  size_t qkv_elems = (size_t)M * 3 * C;
  unsigned short* v2  = qkv + qkv_elems;                    // [B,H,N,D] bf16
  size_t v2_elems  = (size_t)B * H * N * D;
  unsigned short* ctx = v2 + v2_elems;                      // [M, C] bf16

  // 1) qkv = x @ Wqkv
  dim3 g1((M + 127) / 128, (3 * C) / 128);
  gemm_bf16_wmma<false, false><<<g1, 256, 0, stream>>>(x, Wqkv, nullptr, qkv, M, C, 3 * C);

  // 2) v2 = v - s @ v
  dim3 g2((N + 127) / 128, B * H);
  v2_kernel<<<g2, 256, 0, stream>>>(s, qkv, v2);

  // 3) attention (softmax(QK^T*scale) @ v2), merge heads
  dim3 g3((N + 15) / 16, B, H);
  attn_kernel<<<g3, 128, 0, stream>>>(qkv, v2, ctx);

  // 4) out = ctx @ Wproj + bproj (fp32 out)
  dim3 g4((M + 127) / 128, C / 128);
  gemm_bf16_wmma<true, true><<<g4, 256, 0, stream>>>(ctx, Wproj, bproj, (float*)d_out, M, C, C);
}
